// NTXentLoss_9448928051681
// MI455X (gfx1250) — compile-verified, matrix operand order checked by reading, not run on hardware
//
#include <hip/hip_runtime.h>
#include <hip/hip_bf16.h>

typedef __attribute__((ext_vector_type(16))) __bf16 v16bf;
typedef __attribute__((ext_vector_type(8)))  float  v8f;

#define B_ROWS 4096
#define D      256
#define NROWS  8192          // 2B
#define TEMP_INV 2.0f        // 1/0.5
#define LOG2E 1.44269504088896340736f
#define LN2   0.69314718055994530942f

#define LROW 33              // padded LDS row stride in uint4 (528 B)

union Frag { uint4 q[2]; v16bf v; };

// --------- Stage 1: row-normalize, emit bf16 fN + reciprocal norms ----------
__global__ void __launch_bounds__(256) ntxent_normalize(
    const float* __restrict__ f1, const float* __restrict__ f2,
    __bf16* __restrict__ fN, float* __restrict__ rnorm)
{
  int wave = threadIdx.x >> 5;
  int lane = threadIdx.x & 31;
  int row  = blockIdx.x * 8 + wave;
  const float* src = (row < B_ROWS) ? (f1 + (size_t)row * D)
                                    : (f2 + (size_t)(row - B_ROWS) * D);
  const float4* s4 = (const float4*)src;
  float4 a = s4[lane];
  float4 b = s4[lane + 32];
  float ss = a.x*a.x + a.y*a.y + a.z*a.z + a.w*a.w
           + b.x*b.x + b.y*b.y + b.z*b.z + b.w*b.w;
#pragma unroll
  for (int m = 16; m >= 1; m >>= 1) ss += __shfl_xor(ss, m, 32);
  float rn = 1.0f / sqrtf(fmaxf(ss, 1e-24f));
  if (lane == 0) rnorm[row] = rn;

  union { __bf16 h[4]; uint2 u; } p0, p1;
  p0.h[0] = (__bf16)(a.x*rn); p0.h[1] = (__bf16)(a.y*rn);
  p0.h[2] = (__bf16)(a.z*rn); p0.h[3] = (__bf16)(a.w*rn);
  p1.h[0] = (__bf16)(b.x*rn); p1.h[1] = (__bf16)(b.y*rn);
  p1.h[2] = (__bf16)(b.z*rn); p1.h[3] = (__bf16)(b.w*rn);
  uint2* dst = (uint2*)(fN + (size_t)row * D);
  dst[lane]      = p0.u;
  dst[lane + 32] = p1.u;
}

// --------- Stage 2: fused bf16-WMMA Gram GEMM + streaming logsumexp --------
// Grid: 64 WGs x 256 thr (8 waves). Wave w owns M-tile (blockIdx*8+w)*16 rows.
// Double-buffered 32-column super-tiles (one barrier per 16 WMMAs), two
// independent accumulator chains, and B fragments software-pipelined one
// kc-step ahead so ds_load_b128 latency overlaps a full WMMA pair.
__global__ void __launch_bounds__(256, 1) ntxent_gemm_lse(
    const __bf16* __restrict__ fN, float* __restrict__ lse)
{
  __shared__ uint4 tileB[2][32 * LROW];   // 2 x 32 rows x 528 B

  const int wave = threadIdx.x >> 5;
  const int lane = threadIdx.x & 31;
  const int hl   = lane >> 4;                 // which 16-lane half
  const int mt   = blockIdx.x * 8 + wave;     // M-tile index (0..511)
  const int rowA = mt * 16 + (lane & 15);

  const uint4* gA = (const uint4*)fN;         // 32 uint4 per 256-elem bf16 row

  // Resident A fragments: 16x256 bf16 per wave (ISA 16-bit A layout:
  // lanes 0-15 hold K pairs {half0}, lanes 16-31 {half1}; 2 x b128 per frag).
  Frag a[8];
#pragma unroll
  for (int kc = 0; kc < 8; ++kc) {
    size_t base = (size_t)rowA * 32 + kc * 4 + hl;
    a[kc].q[0] = gA[base];
    a[kc].q[1] = gA[base + 2];
  }

  float sum[8];
#pragma unroll
  for (int v = 0; v < 8; ++v) sum[v] = 0.0f;

  const float K1 = TEMP_INV * LOG2E;          // exp(logit-2)=exp2(K1*(dot-1))

  // Staging: 256 threads move 32 rows x 512 B (16 KB); 8 threads/row, 64 B ea.
  const int srow = threadIdx.x >> 3;          // 0..31
  const int sseg = threadIdx.x & 7;           // 0..7 (64 B segments)
  const int base0 = (lane & 15) * LROW + hl;          // N-tile 0 fragment base
  const int base1 = ((lane & 15) + 16) * LROW + hl;   // N-tile 1 fragment base

  // Prologue: stage super-tile 0 into buffer 0.
  {
    size_t gb = (size_t)srow * 32 + sseg * 4;
#pragma unroll
    for (int k = 0; k < 4; ++k)
      tileB[0][srow * LROW + sseg * 4 + k] = gA[gb + k];
  }

  for (int st = 0; st < 256; ++st) {
    __syncthreads();   // staging of buf[st&1] visible; prior compute done

    // Stage next super-tile (mod-wrap keeps control flow uniform).
    {
      int nst = (st + 1) & 255;
      size_t gb = (size_t)(nst * 32 + srow) * 32 + sseg * 4;
      uint4* buf = tileB[(st + 1) & 1];
#pragma unroll
      for (int k = 0; k < 4; ++k)
        buf[srow * LROW + sseg * 4 + k] = gA[gb + k];
    }

    // Compute both 16-col N-tiles of current super-tile: two WMMA chains,
    // with B fragments prefetched one kc-step ahead.
    const uint4* cur = tileB[st & 1];
    v8f c0 = {0.f,0.f,0.f,0.f,0.f,0.f,0.f,0.f};
    v8f c1 = {0.f,0.f,0.f,0.f,0.f,0.f,0.f,0.f};

    Frag b0, b1, nb0, nb1;
    b0.q[0] = cur[base0];
    b0.q[1] = cur[base0 + 2];
    b1.q[0] = cur[base1];
    b1.q[1] = cur[base1 + 2];
#pragma unroll
    for (int kc = 0; kc < 8; ++kc) {
      if (kc < 7) {
        int o = (kc + 1) * 4;
        nb0.q[0] = cur[base0 + o];
        nb0.q[1] = cur[base0 + o + 2];
        nb1.q[0] = cur[base1 + o];
        nb1.q[1] = cur[base1 + o + 2];
      }
      c0 = __builtin_amdgcn_wmma_f32_16x16x32_bf16(
             false, a[kc].v, false, b0.v, (short)0, c0, false, false);
      c1 = __builtin_amdgcn_wmma_f32_16x16x32_bf16(
             false, a[kc].v, false, b1.v, (short)0, c1, false, false);
      b0 = nb0;
      b1 = nb1;
    }
#pragma unroll
    for (int v = 0; v < 8; ++v)
      sum[v] += __builtin_amdgcn_exp2f(fmaf(c0[v], K1, -K1))
              + __builtin_amdgcn_exp2f(fmaf(c1[v], K1, -K1));
  }

  // Reduce each row sum across its 16-lane half (C layout: VGPR v, half h
  // holds row M = h*8 + v, column N = lane&15).
#pragma unroll
  for (int v = 0; v < 8; ++v) {
#pragma unroll
    for (int m = 1; m <= 8; m <<= 1) sum[v] += __shfl_xor(sum[v], m, 32);
  }
  int l4 = lane & 15;
  if (l4 < 8) {
    float s = sum[0];
#pragma unroll
    for (int v = 1; v < 8; ++v) s = (l4 == v) ? sum[v] : s;
    int row = mt * 16 + hl * 8 + l4;
    lse[row] = 2.0f + LN2 * __builtin_amdgcn_logf(s);
  }
}

// --------- Stage 3: per-row (lse - picked) with exact f32 picked logit ------
__global__ void __launch_bounds__(256) ntxent_perrow(
    const float* __restrict__ f1, const float* __restrict__ f2,
    const float* __restrict__ rnorm, const float* __restrict__ lse,
    float* __restrict__ perrow)
{
  int wave = threadIdx.x >> 5;
  int lane = threadIdx.x & 31;
  int i = blockIdx.x * 8 + wave;
  int j = (i < B_ROWS) ? i : i - B_ROWS;
  const float4* pi = (const float4*)((i < B_ROWS) ? f1 + (size_t)i * D
                                                  : f2 + (size_t)(i - B_ROWS) * D);
  const float4* pj = (const float4*)(f1 + (size_t)j * D);
  float4 x0 = pi[lane], y0 = pj[lane];
  float4 x1 = pi[lane + 32], y1 = pj[lane + 32];
  float d = 0.0f;
  d = fmaf(x0.x, y0.x, d); d = fmaf(x0.y, y0.y, d);
  d = fmaf(x0.z, y0.z, d); d = fmaf(x0.w, y0.w, d);
  d = fmaf(x1.x, y1.x, d); d = fmaf(x1.y, y1.y, d);
  d = fmaf(x1.z, y1.z, d); d = fmaf(x1.w, y1.w, d);
#pragma unroll
  for (int m = 16; m >= 1; m >>= 1) d += __shfl_xor(d, m, 32);
  if (lane == 0) {
    float picked = TEMP_INV * rnorm[i] * rnorm[j] * d;
    perrow[i] = lse[i] - picked;
  }
}

// --------- Stage 4: deterministic mean over 8192 rows -----------------------
__global__ void __launch_bounds__(256) ntxent_reduce(
    const float* __restrict__ perrow, float* __restrict__ out)
{
  __shared__ float part[8];
  int t = threadIdx.x;
  float s = 0.0f;
  for (int k = t; k < NROWS; k += 256) s += perrow[k];
#pragma unroll
  for (int m = 16; m >= 1; m >>= 1) s += __shfl_xor(s, m, 32);
  if ((t & 31) == 0) part[t >> 5] = s;
  __syncthreads();
  if (t == 0) {
    float tot = 0.0f;
#pragma unroll
    for (int w = 0; w < 8; ++w) tot += part[w];
    out[0] = tot / (float)NROWS;
  }
}

extern "C" void kernel_launch(void* const* d_in, const int* in_sizes, int n_in,
                              void* d_out, int out_size, void* d_ws, size_t ws_size,
                              hipStream_t stream) {
  const float* f1 = (const float*)d_in[0];
  const float* f2 = (const float*)d_in[1];
  char* ws = (char*)d_ws;
  __bf16* fN    = (__bf16*)ws;                                  // 4 MB
  float*  rnorm = (float*)(ws + 4u*1024*1024);                  // 32 KB
  float*  lse   = (float*)(ws + 4u*1024*1024 + 32u*1024);       // 32 KB
  float*  prow  = (float*)(ws + 4u*1024*1024 + 64u*1024);       // 32 KB

  ntxent_normalize<<<NROWS/8, 256, 0, stream>>>(f1, f2, fN, rnorm);
  ntxent_gemm_lse<<<NROWS/16/8, 256, 0, stream>>>(fN, lse);
  ntxent_perrow  <<<NROWS/8, 256, 0, stream>>>(f1, f2, rnorm, lse, prow);
  ntxent_reduce  <<<1, 256, 0, stream>>>(prow, (float*)d_out);
}
